// AttentionBlock_10814727651793
// MI455X (gfx1250) — matrix operand order repaired and checked by reading, not compiled, not run on hardware
//
#include <hip/hip_runtime.h>
#include <cstdint>
#include <cstddef>

// ---------------------------------------------------------------------------
// AttentionBlock (GroupNorm -> QKV -> softmax(QK^T/sqrt(c))V -> proj -> +res)
// b=4, c=256, h=w=64 -> n=4096. Flash-attention style, bf16 WMMA GEMMs,
// fp32 softmax state / accumulators. gfx1250 wave32.
// Round 2: K/V tiles staged with GLOBAL_LOAD_ASYNC_TO_LDS_B128 (ASYNCcnt),
// double-buffered so the async engine overlaps WMMA compute.
// ---------------------------------------------------------------------------

typedef __attribute__((ext_vector_type(16))) __bf16 v16bf;
typedef __attribute__((ext_vector_type(8)))  float  v8f;

#define B_ 4
#define C_ 256
#define N_ 4096
#define LOG2E 1.4426950408889634f

union BfVec { uint32_t u[8]; v16bf v; };
struct alignas(16) Bf8 { __bf16 h[8]; };

// A-operand (16x32 bf16) K-offset for VGPR v, lane-half hi:
//   K = 2v + h + 16*hi
__device__ __forceinline__ int koffA(int v, int hi) {
    return 2 * v + 16 * hi;
}

// Async 16-byte copy global -> LDS (no VGPR data path, tracked by ASYNCcnt).
// VDST = wave-relative LDS byte address (low 32 bits of generic LDS pointer),
// VADDR = 64-bit global address.
__device__ __forceinline__ void async_ld16(void* lds, const void* g) {
    uint32_t l = (uint32_t)(uintptr_t)lds;
    uint64_t a = (uint64_t)(uintptr_t)g;
    asm volatile("global_load_async_to_lds_b128 %0, %1, off"
                 :: "v"(l), "v"(a) : "memory");
}
__device__ __forceinline__ void wait_async0() {
    asm volatile("s_wait_asynccnt 0" ::: "memory");
}

// ---------------------------------------------------------------------------
// 1) GroupNorm statistics: one block per (batch, group)
// ---------------------------------------------------------------------------
__global__ void gn_stats(const float* __restrict__ x, float* __restrict__ stats) {
    __shared__ float ss[256], sq[256];
    int bg = blockIdx.x;
    int b = bg >> 3, g = bg & 7;
    const float* xp = x + ((size_t)(b * C_) + g * 32) * N_;
    float s = 0.f, q = 0.f;
    for (int i = threadIdx.x; i < 32 * N_; i += 256) {
        float v = xp[i];
        s += v; q += v * v;
    }
    ss[threadIdx.x] = s; sq[threadIdx.x] = q;
    __syncthreads();
    for (int off = 128; off > 0; off >>= 1) {
        if ((int)threadIdx.x < off) {
            ss[threadIdx.x] += ss[threadIdx.x + off];
            sq[threadIdx.x] += sq[threadIdx.x + off];
        }
        __syncthreads();
    }
    if (threadIdx.x == 0) {
        float inv_n = 1.0f / (32.0f * N_);
        float mean = ss[0] * inv_n;
        float var  = sq[0] * inv_n - mean * mean;
        stats[bg * 2]     = mean;
        stats[bg * 2 + 1] = rsqrtf(var + 1e-5f);
    }
}

// ---------------------------------------------------------------------------
// 2a) Convert GEMM weights to bf16
// ---------------------------------------------------------------------------
__global__ void wconv(const float* __restrict__ wq, const float* __restrict__ wp,
                      __bf16* __restrict__ w2, __bf16* __restrict__ wp2) {
    int i = blockIdx.x * 256 + threadIdx.x;
    if (i < 768 * 256) w2[i]  = (__bf16)wq[i];
    if (i < 256 * 256) wp2[i] = (__bf16)wp[i];
}

// ---------------------------------------------------------------------------
// 2b) Apply GroupNorm and transpose to XN^T[b][p][c] bf16 (LDS-tiled)
// ---------------------------------------------------------------------------
__global__ void gn_apply(const float* __restrict__ x, const float* __restrict__ stats,
                         const float* __restrict__ gamma, const float* __restrict__ beta,
                         __bf16* __restrict__ xnT) {
    __shared__ float tile[32][65];
    int blk = blockIdx.x;
    int b = blk >> 9; int rem = blk & 511;
    int ct = rem >> 6; int pt = rem & 63;          // c-tile of 32 (== one group), p-tile of 64
    int c0 = ct * 32, p0 = pt * 64;
    float mean = stats[(b * 8 + ct) * 2];
    float rstd = stats[(b * 8 + ct) * 2 + 1];
    for (int e = threadIdx.x; e < 2048; e += 256) {
        int cc = e >> 6, pp = e & 63;
        int c = c0 + cc;
        float v = x[((size_t)(b * C_) + c) * N_ + p0 + pp];
        tile[cc][pp] = (v - mean) * rstd * gamma[c] + beta[c];
    }
    __syncthreads();
    for (int e = threadIdx.x; e < 2048; e += 256) {
        int pp = e >> 5, cc = e & 31;
        xnT[((size_t)b * N_ + p0 + pp) * 256 + c0 + cc] = (__bf16)tile[cc][pp];
    }
}

// ---------------------------------------------------------------------------
// 3) QKV GEMM: qkv[o][p] = sum_c W[o][c] * xn[p][c] + bias[o]
//    One wave per 16(o) x 16(p) tile; K=256 via 8x v_wmma_f32_16x16x32_bf16.
//    Q -> qT[b][p][c] (scaled by 1/16), K -> kT[b][p][c], V -> vC[b][c][p].
// ---------------------------------------------------------------------------
__global__ void __launch_bounds__(256) qkv_gemm(
    const __bf16* __restrict__ w2, const __bf16* __restrict__ xnT,
    const float* __restrict__ bq,
    __bf16* __restrict__ qT, __bf16* __restrict__ kT, __bf16* __restrict__ vC) {
    int wid  = blockIdx.x * 8 + (threadIdx.x >> 5);
    int lane = threadIdx.x & 31, lm = lane & 15, hi = lane >> 4;
    int b = wid / 12288; int rem = wid % 12288;            // 48 o-tiles * 256 p-tiles
    int o0 = (rem >> 8) * 16; int p0 = (rem & 255) * 16;
    const __bf16* xb = xnT + (size_t)b * N_ * C_;
    v8f acc = {};
    for (int kc = 0; kc < 256; kc += 32) {
        BfVec A, Bv;
        const __bf16* ap = w2 + (o0 + lm) * 256 + kc;
        #pragma unroll
        for (int v = 0; v < 8; ++v)
            A.u[v] = *(const uint32_t*)(ap + koffA(v, hi));
        const __bf16* bp = xb + (size_t)(p0 + lm) * 256 + kc + 16 * hi;
        ((uint4*)Bv.u)[0] = *(const uint4*)(bp);
        ((uint4*)Bv.u)[1] = *(const uint4*)(bp + 8);
        acc = __builtin_amdgcn_wmma_f32_16x16x32_bf16(false, A.v, false, Bv.v,
                                                      (short)0, acc, false, false);
    }
    if (o0 < 512) {                                   // Q or K: [p][c] layout, packed store
        float scale = (o0 < 256) ? 0.0625f : 1.0f;    // fold 1/sqrt(256) into Q
        __bf16* dst = (o0 < 256) ? qT : kT;
        int ob = (o0 < 256) ? o0 : (o0 - 256);
        Bf8 pk;
        #pragma unroll
        for (int r = 0; r < 8; ++r) {
            int orow = o0 + r + 8 * hi;
            pk.h[r] = (__bf16)((acc[r] + bq[orow]) * scale);
        }
        *(Bf8*)(dst + ((size_t)b * N_ + p0 + lm) * 256 + ob + 8 * hi) = pk;
    } else {                                          // V: [c][p] layout
        #pragma unroll
        for (int r = 0; r < 8; ++r) {
            int orow = o0 + r + 8 * hi;
            vC[((size_t)b * C_ + (orow - 512)) * N_ + p0 + lm] = (__bf16)(acc[r] + bq[orow]);
        }
    }
}

// ---------------------------------------------------------------------------
// 4) Flash attention. Block = 16 query rows, 4 waves (each owns 64 V-columns).
//    Online softmax, bf16 WMMA for S = Q K^T and O += P V.
//    K/V tiles double-buffered via async global->LDS DMA (ASYNCcnt).
// ---------------------------------------------------------------------------
__global__ void __launch_bounds__(128) flash_attn(
    const __bf16* __restrict__ qT, const __bf16* __restrict__ kT,
    const __bf16* __restrict__ vC, __bf16* __restrict__ ao) {
    __shared__ alignas(16) __bf16 qt[16][256];      // Q rows  [i][c]
    __shared__ alignas(16) __bf16 kt[2][32][256];   // K tile  [j][c], double buffered
    __shared__ alignas(16) __bf16 vt[2][256][32];   // V tile  [c][j], double buffered
    __shared__ alignas(16) __bf16 ps[4][16][32];    // per-wave P transpose scratch
    int tid = threadIdx.x;
    int wave = tid >> 5, lane = tid & 31, lm = lane & 15, hi = lane >> 4;
    int b = blockIdx.x >> 8; int i0 = (blockIdx.x & 255) * 16;
    const __bf16* qb = qT + ((size_t)b * N_ + i0) * 256;
    const __bf16* kb = kT + (size_t)b * N_ * 256;
    const __bf16* vb = vC + (size_t)b * C_ * N_;

    // Prologue: async-stage Q tile and the first K/V tiles into buffer 0.
    for (int t = tid; t < 512; t += 128)
        async_ld16(&qt[t >> 5][(t & 31) * 8], qb + (t >> 5) * 256 + (t & 31) * 8);
    for (int t = tid; t < 1024; t += 128) {
        async_ld16(&kt[0][t >> 5][(t & 31) * 8],
                   kb + (size_t)(t >> 5) * 256 + (t & 31) * 8);
        async_ld16(&vt[0][t >> 2][(t & 3) * 8],
                   vb + (size_t)(t >> 2) * N_ + (t & 3) * 8);
    }

    float m[8], l[8];
    v8f O[4];
    #pragma unroll
    for (int r = 0; r < 8; ++r) { m[r] = -1e30f; l[r] = 0.f; }
    #pragma unroll
    for (int t = 0; t < 4; ++t) O[t] = (v8f){};
    int c0w = wave * 64;

    for (int jt = 0; jt < N_ / 32; ++jt) {
        int cur = jt & 1;
        // Own async ops complete, then barrier => everyone's transfers landed.
        wait_async0();
        __syncthreads();
        if (jt + 1 < N_ / 32) {     // kick off next tile into the other buffer
            int jn = (jt + 1) * 32;
            for (int t = tid; t < 1024; t += 128) {
                async_ld16(&kt[cur ^ 1][t >> 5][(t & 31) * 8],
                           kb + (size_t)(jn + (t >> 5)) * 256 + (t & 31) * 8);
                async_ld16(&vt[cur ^ 1][t >> 2][(t & 3) * 8],
                           vb + (size_t)(t >> 2) * N_ + jn + (t & 3) * 8);
            }
        }

        // S tile (16 x 32): two 16x16 WMMA accumulators over 8 K-chunks
        v8f s0 = {}, s1 = {};
        #pragma unroll
        for (int kc = 0; kc < 256; kc += 32) {
            BfVec A, B0, B1;
            #pragma unroll
            for (int v = 0; v < 8; ++v)
                A.u[v] = *(const uint32_t*)&qt[lm][kc + koffA(v, hi)];
            ((uint4*)B0.u)[0] = *(const uint4*)&kt[cur][lm][kc + 16 * hi];
            ((uint4*)B0.u)[1] = *(const uint4*)&kt[cur][lm][kc + 16 * hi + 8];
            ((uint4*)B1.u)[0] = *(const uint4*)&kt[cur][16 + lm][kc + 16 * hi];
            ((uint4*)B1.u)[1] = *(const uint4*)&kt[cur][16 + lm][kc + 16 * hi + 8];
            s0 = __builtin_amdgcn_wmma_f32_16x16x32_bf16(false, A.v, false, B0.v,
                                                         (short)0, s0, false, false);
            s1 = __builtin_amdgcn_wmma_f32_16x16x32_bf16(false, A.v, false, B1.v,
                                                         (short)0, s1, false, false);
        }

        // Online softmax update (row reductions across 16-lane halves, wave32)
        float p0v[8], p1v[8], al[8];
        #pragma unroll
        for (int r = 0; r < 8; ++r) {
            float vm = fmaxf(s0[r], s1[r]);
            #pragma unroll
            for (int off = 8; off >= 1; off >>= 1)
                vm = fmaxf(vm, __shfl_xor(vm, off, 32));
            float mn = fmaxf(m[r], vm);
            al[r]  = exp2f((m[r] - mn) * LOG2E);
            p0v[r] = exp2f((s0[r] - mn) * LOG2E);
            p1v[r] = exp2f((s1[r] - mn) * LOG2E);
            float rs = p0v[r] + p1v[r];
            #pragma unroll
            for (int off = 8; off >= 1; off >>= 1)
                rs += __shfl_xor(rs, off, 32);
            l[r] = l[r] * al[r] + rs;
            m[r] = mn;
        }
        #pragma unroll
        for (int t = 0; t < 4; ++t)
            #pragma unroll
            for (int r = 0; r < 8; ++r)
                O[t][r] *= al[r];

        // Route P (C-layout) -> A-layout through per-wave LDS scratch
        #pragma unroll
        for (int r = 0; r < 8; ++r) {
            ps[wave][r + 8 * hi][lm]      = (__bf16)p0v[r];
            ps[wave][r + 8 * hi][16 + lm] = (__bf16)p1v[r];
        }
        asm volatile("s_wait_dscnt 0" ::: "memory");
        BfVec Pa;
        #pragma unroll
        for (int v = 0; v < 8; ++v)
            Pa.u[v] = *(const uint32_t*)&ps[wave][lm][koffA(v, hi)];

        // O[16 x 64] += P[16 x 32] * V[32 x 64]  (4 N-tiles)
        #pragma unroll
        for (int t = 0; t < 4; ++t) {
            BfVec Bv;
            ((uint4*)Bv.u)[0] = *(const uint4*)&vt[cur][c0w + t * 16 + lm][16 * hi];
            ((uint4*)Bv.u)[1] = *(const uint4*)&vt[cur][c0w + t * 16 + lm][16 * hi + 8];
            O[t] = __builtin_amdgcn_wmma_f32_16x16x32_bf16(false, Pa.v, false, Bv.v,
                                                           (short)0, O[t], false, false);
        }
    }

    #pragma unroll
    for (int r = 0; r < 8; ++r) {
        float inv = 1.0f / l[r];
        #pragma unroll
        for (int t = 0; t < 4; ++t)
            ao[((size_t)b * N_ + i0 + r + 8 * hi) * 256 + c0w + t * 16 + lm] =
                (__bf16)(O[t][r] * inv);
    }
}

// ---------------------------------------------------------------------------
// 5) Output projection + bias + residual -> fp32 out
// ---------------------------------------------------------------------------
__global__ void __launch_bounds__(256) proj_gemm(
    const __bf16* __restrict__ wp2, const __bf16* __restrict__ ao,
    const float* __restrict__ bp, const float* __restrict__ x,
    float* __restrict__ out) {
    int wid  = blockIdx.x * 8 + (threadIdx.x >> 5);
    int lane = threadIdx.x & 31, lm = lane & 15, hi = lane >> 4;
    int b = wid / 4096; int rem = wid % 4096;              // 16 o-tiles * 256 p-tiles
    int o0 = (rem >> 8) * 16; int p0 = (rem & 255) * 16;
    const __bf16* ab = ao + (size_t)b * N_ * 256;
    v8f acc = {};
    for (int kc = 0; kc < 256; kc += 32) {
        BfVec A, Bv;
        const __bf16* ap = wp2 + (o0 + lm) * 256 + kc;
        #pragma unroll
        for (int v = 0; v < 8; ++v)
            A.u[v] = *(const uint32_t*)(ap + koffA(v, hi));
        const __bf16* bpt = ab + (size_t)(p0 + lm) * 256 + kc + 16 * hi;
        ((uint4*)Bv.u)[0] = *(const uint4*)(bpt);
        ((uint4*)Bv.u)[1] = *(const uint4*)(bpt + 8);
        acc = __builtin_amdgcn_wmma_f32_16x16x32_bf16(false, A.v, false, Bv.v,
                                                      (short)0, acc, false, false);
    }
    #pragma unroll
    for (int r = 0; r < 8; ++r) {
        int orow = o0 + r + 8 * hi;
        size_t idx = ((size_t)b * C_ + orow) * N_ + p0 + lm;
        out[idx] = acc[r] + bp[orow] + x[idx];
    }
}

// ---------------------------------------------------------------------------
extern "C" void kernel_launch(void* const* d_in, const int* in_sizes, int n_in,
                              void* d_out, int out_size, void* d_ws, size_t ws_size,
                              hipStream_t stream) {
    const float* x      = (const float*)d_in[0];
    const float* gamma  = (const float*)d_in[1];
    const float* beta   = (const float*)d_in[2];
    const float* w_qkv  = (const float*)d_in[3];
    const float* b_qkv  = (const float*)d_in[4];
    const float* w_proj = (const float*)d_in[5];
    const float* b_proj = (const float*)d_in[6];
    float* out = (float*)d_out;

    char* ws = (char*)d_ws;
    float*  stats = (float*)ws;  ws += 512;
    __bf16* w2    = (__bf16*)ws; ws += (size_t)768 * 256 * 2;
    __bf16* wp2   = (__bf16*)ws; ws += (size_t)256 * 256 * 2;
    const size_t planeB = (size_t)B_ * N_ * C_ * 2;   // 8.4 MB each
    __bf16* xnT = (__bf16*)ws; ws += planeB;
    __bf16* qT  = (__bf16*)ws; ws += planeB;
    __bf16* kT  = (__bf16*)ws; ws += planeB;
    __bf16* vC  = (__bf16*)ws; ws += planeB;
    __bf16* ao  = (__bf16*)ws; ws += planeB;

    gn_stats <<<32,   256, 0, stream>>>(x, stats);
    wconv    <<<768,  256, 0, stream>>>(w_qkv, w_proj, w2, wp2);
    gn_apply <<<2048, 256, 0, stream>>>(x, stats, gamma, beta, xnT);
    qkv_gemm <<<6144, 256, 0, stream>>>(w2, xnT, b_qkv, qT, kT, vC);
    flash_attn<<<1024, 128, 0, stream>>>(qT, kT, vC, ao);
    proj_gemm<<<2048, 256, 0, stream>>>(wp2, ao, b_proj, x, out);
}
